// Encoder_pMHC_Bilstm_40544491274774
// MI455X (gfx1250) — compile-verified, matrix operand order checked by reading, not run on hardware
//
#include <hip/hip_runtime.h>
#include <hip/hip_bf16.h>

typedef __attribute__((ext_vector_type(16))) _Float16 v16h;
typedef __attribute__((ext_vector_type(8)))  _Float16 v8h;
typedef __attribute__((ext_vector_type(8)))  float    v8f;

#define BB 16384
#define TT 48
#define II 5
#define HH 64
#define GG 256   // 4*H

// packed weight layout (halves):
//   M0 (layer0: Whh0^T for kc=0,1 ; Wih0^T zero-padded for kc=2): 3*16*32*16 = 24576
//   M1 (Wih1^T): 2*16*32*16 = 16384   at offset 24576
//   M2 (Whh1^T): 16384                at offset 40960
// then (bytes) b0[256] f32 at 114688, b1[256] f32 after.
#define PK_M1 24576
#define PK_M2 40960
#define PK_HALVES 57344
#define B0_BYTE_OFF 114688

__device__ __forceinline__ int kmap(int e, int half) {
    int p = e >> 1;
    int base = (p < 4) ? (2 * p) : (16 + 2 * (p - 4));
    return base + 8 * half + (e & 1);
}

// ---------------- pack kernel: f32 weights -> f16 WMMA B-fragments ----------------
__global__ __launch_bounds__(256) void pack_kernel(
    const float* __restrict__ wih0, const float* __restrict__ whh0,
    const float* __restrict__ wih1, const float* __restrict__ whh1,
    const float* __restrict__ bih0, const float* __restrict__ bhh0,
    const float* __restrict__ bih1, const float* __restrict__ bhh1,
    _Float16* __restrict__ pk, float* __restrict__ b0, float* __restrict__ b1)
{
    int gid = blockIdx.x * 256 + threadIdx.x;
    if (gid < PK_HALVES) {
        int mtx, rel;
        if (gid < PK_M1)      { mtx = 0; rel = gid; }
        else if (gid < PK_M2) { mtx = 1; rel = gid - PK_M1; }
        else                  { mtx = 2; rel = gid - PK_M2; }
        int e    = rel & 15;
        int lane = (rel >> 4) & 31;
        int frag = rel >> 9;
        int nt   = frag & 15;
        int kc   = frag >> 4;
        int half = lane >> 4;
        int n    = nt * 16 + (lane & 15);
        int kl   = kmap(e, half);
        float v;
        if (mtx == 0) {
            if (kc < 2) v = whh0[n * HH + kc * 32 + kl];
            else        v = (kl < II) ? wih0[n * II + kl] : 0.0f;
        } else if (mtx == 1) {
            v = wih1[n * HH + kc * 32 + kl];
        } else {
            v = whh1[n * HH + kc * 32 + kl];
        }
        pk[gid] = (_Float16)v;
    } else if (gid < PK_HALVES + 512) {
        int r = gid - PK_HALVES;
        if (r < 256) b0[r] = bih0[r] + bhh0[r];
        else         b1[r - 256] = bih1[r - 256] + bhh1[r - 256];
    }
}

// ---------------- fragment loaders ----------------
__device__ __forceinline__ v16h load_w_frag(const _Float16* __restrict__ p) {
    v8h lo = *(const v8h*)p;
    v8h hi = *(const v8h*)(p + 8);
    v16h a;
#pragma unroll
    for (int i = 0; i < 8; i++) { a[i] = lo[i]; a[i + 8] = hi[i]; }
    return a;
}

// A matrix fragment from row-major LDS [16][stride] f16 (ISA 16-bit A layout)
__device__ __forceinline__ v16h load_a_frag(const _Float16* base, int stride, int lane, int kc) {
    int row = lane & 15, half = lane >> 4;
    const _Float16* rp = base + row * stride + kc * 32 + 8 * half;
    v8h lo = *(const v8h*)rp;
    v8h hi = *(const v8h*)(rp + 16);
    v16h a;
#pragma unroll
    for (int i = 0; i < 8; i++) { a[i] = lo[i]; a[i + 8] = hi[i]; }
    return a;
}

__device__ __forceinline__ v8f wmma16(v16h a, v16h b, v8f c) {
    return __builtin_amdgcn_wmma_f32_16x16x32_f16(false, a, false, b, (short)0, c, false, false);
}

__device__ __forceinline__ float fsigmoid(float x) {
    return 1.0f / (1.0f + __expf(-x));
}
__device__ __forceinline__ float ftanh(float x) {
    x = fminf(fmaxf(x, -15.0f), 15.0f);
    float e = __expf(2.0f * x);
    return (e - 1.0f) / (e + 1.0f);
}

// ---------------- fused 2-layer LSTM + LayerNorm + mask ----------------
__global__ __launch_bounds__(256) void lstm_fused_kernel(
    const float* __restrict__ x,
    const _Float16* __restrict__ pk,
    const float* __restrict__ b0g, const float* __restrict__ b1g,
    const float* __restrict__ gamma, const float* __restrict__ beta,
    float* __restrict__ y, float* __restrict__ mask)
{
    __shared__ __attribute__((aligned(16))) _Float16 sh_h0[16 * HH];  // hprev0 / h1_t
    __shared__ __attribute__((aligned(16))) _Float16 sh_h1[16 * HH];  // hprev1
    __shared__ __attribute__((aligned(16))) _Float16 sh_xc[16 * 32];  // x_t zero-padded to K=32
    __shared__ float sh_g[16 * GG];                                   // gate staging
    __shared__ float sh_x[16 * TT * II];                              // x tile
    __shared__ float sh_b0[GG], sh_b1[GG];
    __shared__ float sh_gam[HH], sh_bet[HH];

    const int tid  = threadIdx.x;
    const int wv   = tid >> 5;
    const int lane = tid & 31;
    const int half = lane >> 4;
    const int cl   = lane & 15;

    // ---- resident weight fragments (registers, whole kernel) ----
    v16h w0[2][3], wi1[2][2], wh1[2][2];
#pragma unroll
    for (int n = 0; n < 2; n++) {
        int nt = 2 * wv + n;
#pragma unroll
        for (int kc = 0; kc < 3; kc++)
            w0[n][kc] = load_w_frag(pk + ((kc * 16 + nt) * 32 + lane) * 16);
#pragma unroll
        for (int kc = 0; kc < 2; kc++) {
            wi1[n][kc] = load_w_frag(pk + PK_M1 + ((kc * 16 + nt) * 32 + lane) * 16);
            wh1[n][kc] = load_w_frag(pk + PK_M2 + ((kc * 16 + nt) * 32 + lane) * 16);
        }
    }

    // ---- stage x tile (contiguous: 16 rows of this block), biases, LN params ----
    const int xbase = blockIdx.x * 16 * TT * II;
    for (int i = tid; i < 16 * TT * II; i += 256) sh_x[i] = x[xbase + i];
    sh_b0[tid] = b0g[tid];
    sh_b1[tid] = b1g[tid];
    if (tid < HH) { sh_gam[tid] = gamma[tid]; sh_bet[tid] = beta[tid]; }
    for (int i = tid; i < 16 * HH; i += 256) { sh_h0[i] = (_Float16)0.0f; sh_h1[i] = (_Float16)0.0f; }
    for (int i = tid; i < 16 * 32; i += 256) sh_xc[i] = (_Float16)0.0f;
    __syncthreads();

    // ---- mask: (x[b,t,:] == 0).all ----
    for (int r = tid; r < 16 * TT; r += 256) {
        int m = r / TT, t = r % TT;
        bool z = true;
#pragma unroll
        for (int i = 0; i < II; i++) z = z && (sh_x[(m * TT + t) * II + i] == 0.0f);
        mask[(blockIdx.x * 16 + m) * TT + t] = z ? 1.0f : 0.0f;
    }

    // per-thread cell state: thread owns row m=tid>>4, cols j0..j0+3
    const int em = tid >> 4;
    const int j0 = (tid & 15) * 4;
    float c0[4] = {0, 0, 0, 0}, c1[4] = {0, 0, 0, 0};

    for (int t = 0; t < TT; t++) {
        // stage x_t (f16, padded) for the kc=2 WMMA chunk of layer 0
        if (tid < 16 * II) {
            int m = tid / II, i = tid % II;
            sh_xc[m * 32 + i] = (_Float16)sh_x[(m * TT + t) * II + i];
        }
        __syncthreads();

        // ===== layer 0: gates = [h_prev0 | x_t] @ [Whh0^T ; Wih0^T]  (K = 96) =====
        {
            v8f a0 = {}, a1 = {};
#pragma unroll
            for (int kc = 0; kc < 3; kc++) {
                v16h a = (kc < 2) ? load_a_frag(sh_h0, HH, lane, kc)
                                  : load_a_frag(sh_xc, 32, lane, 0);
                a0 = wmma16(a, w0[0][kc], a0);
                a1 = wmma16(a, w0[1][kc], a1);
            }
#pragma unroll
            for (int v = 0; v < 8; v++) {
                sh_g[(v + 8 * half) * GG + (2 * wv + 0) * 16 + cl] = a0[v];
                sh_g[(v + 8 * half) * GG + (2 * wv + 1) * 16 + cl] = a1[v];
            }
        }
        __syncthreads();

        // ===== layer 0 elementwise: i,f,g,o -> c0, h1_t =====
        {
            const float* gr = &sh_g[em * GG];
#pragma unroll
            for (int q = 0; q < 4; q++) {
                int j = j0 + q;
                float ig = fsigmoid(gr[j]        + sh_b0[j]);
                float fg = fsigmoid(gr[64 + j]   + sh_b0[64 + j]);
                float gg = ftanh   (gr[128 + j]  + sh_b0[128 + j]);
                float og = fsigmoid(gr[192 + j]  + sh_b0[192 + j]);
                c0[q] = fg * c0[q] + ig * gg;
                sh_h0[em * HH + j] = (_Float16)(og * ftanh(c0[q]));
            }
        }
        __syncthreads();

        // ===== layer 1: gates = h1_t @ Wih1^T + h_prev1 @ Whh1^T =====
        {
            v8f a0 = {}, a1 = {};
#pragma unroll
            for (int kc = 0; kc < 2; kc++) {
                v16h a = load_a_frag(sh_h0, HH, lane, kc);
                a0 = wmma16(a, wi1[0][kc], a0);
                a1 = wmma16(a, wi1[1][kc], a1);
            }
#pragma unroll
            for (int kc = 0; kc < 2; kc++) {
                v16h a = load_a_frag(sh_h1, HH, lane, kc);
                a0 = wmma16(a, wh1[0][kc], a0);
                a1 = wmma16(a, wh1[1][kc], a1);
            }
#pragma unroll
            for (int v = 0; v < 8; v++) {
                sh_g[(v + 8 * half) * GG + (2 * wv + 0) * 16 + cl] = a0[v];
                sh_g[(v + 8 * half) * GG + (2 * wv + 1) * 16 + cl] = a1[v];
            }
        }
        __syncthreads();

        // ===== layer 1 elementwise + fused LayerNorm + store y =====
        {
            const float* gr = &sh_g[em * GG];
            float h2[4];
#pragma unroll
            for (int q = 0; q < 4; q++) {
                int j = j0 + q;
                float ig = fsigmoid(gr[j]       + sh_b1[j]);
                float fg = fsigmoid(gr[64 + j]  + sh_b1[64 + j]);
                float gg = ftanh   (gr[128 + j] + sh_b1[128 + j]);
                float og = fsigmoid(gr[192 + j] + sh_b1[192 + j]);
                c1[q] = fg * c1[q] + ig * gg;
                h2[q] = og * ftanh(c1[q]);
            }
            // row of 64 owned by 16 contiguous lanes: wave32 butterfly reduce
            float s = h2[0] + h2[1] + h2[2] + h2[3];
#pragma unroll
            for (int off = 1; off < 16; off <<= 1) s += __shfl_xor(s, off, 32);
            float mu = s * (1.0f / 64.0f);
            float vs = 0.0f;
#pragma unroll
            for (int q = 0; q < 4; q++) { float d = h2[q] - mu; vs += d * d; }
#pragma unroll
            for (int off = 1; off < 16; off <<= 1) vs += __shfl_xor(vs, off, 32);
            float inv = rsqrtf(vs * (1.0f / 64.0f) + 1e-5f);

            size_t gb = ((size_t)(blockIdx.x * 16 + em) * TT + t) * HH + j0;
#pragma unroll
            for (int q = 0; q < 4; q++) {
                y[gb + q] = (h2[q] - mu) * inv * sh_gam[j0 + q] + sh_bet[j0 + q];
                sh_h1[em * HH + j0 + q] = (_Float16)h2[q];
            }
        }
        // top-of-loop barrier orders sh_h1/sh_g reuse for the next step
    }
}

extern "C" void kernel_launch(void* const* d_in, const int* in_sizes, int n_in,
                              void* d_out, int out_size, void* d_ws, size_t ws_size,
                              hipStream_t stream) {
    const float* x     = (const float*)d_in[0];
    const float* Wih0  = (const float*)d_in[1];
    const float* Whh0  = (const float*)d_in[2];
    const float* bih0  = (const float*)d_in[3];
    const float* bhh0  = (const float*)d_in[4];
    const float* Wih1  = (const float*)d_in[5];
    const float* Whh1  = (const float*)d_in[6];
    const float* bih1  = (const float*)d_in[7];
    const float* bhh1  = (const float*)d_in[8];
    const float* gamma = (const float*)d_in[9];
    const float* beta  = (const float*)d_in[10];

    _Float16* pk = (_Float16*)d_ws;
    float* b0 = (float*)((char*)d_ws + B0_BYTE_OFF);
    float* b1 = b0 + 256;

    float* y    = (float*)d_out;
    float* mask = y + (size_t)BB * TT * HH;

    pack_kernel<<<(PK_HALVES + 512 + 255) / 256, 256, 0, stream>>>(
        Wih0, Whh0, Wih1, Whh1, bih0, bhh0, bih1, bhh1, pk, b0, b1);

    lstm_fused_kernel<<<BB / 16, 256, 0, stream>>>(
        x, pk, b0, b1, gamma, beta, y, mask);
}
